// AttentionGate_56745107914940
// MI455X (gfx1250) — matrix operand order repaired and checked by reading, not compiled, not run on hardware
//
#include <hip/hip_runtime.h>

// ---------------------------------------------------------------------------
// AttentionGate fused kernel for gfx1250 (MI455X, wave32, WMMA).
// Kernel 1: pack Wx/Wg (fp32) -> bf16 WMMA A-fragments in d_ws (256 KB).
// Kernel 2: per block = 32 pixels (2 N-tiles):
//   stage x,g into LDS as per-lane bf16 B-fragments (fragment-order layout ->
//   ds_load_b128 straight into WMMA operand registers), 32x
//   v_wmma_f32_16x16x32_bf16 (K=2*256), relu+bias, LayerNorm over C,
//   psi dot + sigmoid, gate original fp32 x (L2-hot re-read).
// ---------------------------------------------------------------------------

typedef __attribute__((ext_vector_type(16))) __bf16        v16bf;
typedef __attribute__((ext_vector_type(8)))  float         v8f;
typedef __attribute__((ext_vector_type(8)))  unsigned int  v8u;

#define C_DIM    256
#define HW       4096          // 64*64
#define BSTRIDE  1048576       // 256*4096
#define PIXB     32            // pixels per block (2 N-tiles of 16)
#define NWAVES   16
#define NTHREADS 512

__device__ __forceinline__ unsigned int pk_bf16(float a, float b) {
  // round-to-nearest-even f32 -> bf16, packed (a -> [15:0], b -> [31:16])
  unsigned int ua = __float_as_uint(a);
  unsigned int ub = __float_as_uint(b);
  unsigned int ha = (ua + 0x7FFFu + ((ua >> 16) & 1u)) >> 16;
  unsigned int hb = (ub + 0x7FFFu + ((ub >> 16) & 1u)) >> 16;
  return (hb << 16) | (ha & 0xFFFFu);
}

// ---------------------------------------------------------------------------
// Weight pre-pack: ws[((mat*16+mt)*8+kb)*32*8 + lane*8 + v] =
//   pk_bf16(W[row, k0], W[row, k0+1])
//   row = mt*16 + (lane&15); k0 = kb*32 + ((v&4)?16:0) + (v&3)*2 + (lane>>4)*8
// (exact 16-bit A-matrix 16x32 register layout per CDNA5 ISA 7.12.2)
// Total: 2 * 16 * 8 * 32 * 8 uints = 64K uints = 256 KB.
// ---------------------------------------------------------------------------
__global__ __launch_bounds__(256)
void pack_weights_kernel(const float* __restrict__ Wx,
                         const float* __restrict__ Wg,
                         unsigned int* __restrict__ ws) {
  const int tid  = blockIdx.x * 256 + threadIdx.x;   // 0..65535
  const int v    = tid & 7;
  const int lane = (tid >> 3) & 31;
  const int kb   = (tid >> 8) & 7;
  const int mt   = (tid >> 11) & 15;
  const int mat  = tid >> 15;
  const float* __restrict__ W = mat ? Wg : Wx;
  const int row = mt * 16 + (lane & 15);
  const int k0  = kb * 32 + ((v & 4) ? 16 : 0) + (v & 3) * 2 + ((lane >> 4) << 3);
  const float2 w2 = *(const float2*)(W + row * C_DIM + k0);
  ws[tid] = pk_bf16(w2.x, w2.y);
}

// B-fragment LDS layout: [(nt*8+kb)*2 + vhalf][lane][4 dwords]
//   fragment dword v of lane l for K-block kb, N-tile nt lives at
//   ((((nt*8+kb)*2+(v>>2))*32 + l)*4 + (v&3)
// -> each WMMA B operand = two 16B-aligned ds_load_b128 per lane.
#define BFRAG_U4(nt, kb, vh, lane) ((((nt) * 8 + (kb)) * 2 + (vh)) * 32 + (lane))

union FragU {
  uint4 q[2];
  v8u   v;
};

__global__ __launch_bounds__(NTHREADS, 1)
void attn_gate_kernel(const float* __restrict__ x,
                      const float* __restrict__ g,
                      const unsigned int* __restrict__ ws,   // packed bf16 A-frags
                      const float* __restrict__ Wpsi,
                      const float* __restrict__ ln_gamma,
                      const float* __restrict__ ln_beta,
                      const float* __restrict__ bxg,
                      const float* __restrict__ bpsi,
                      float* __restrict__ out) {
  __shared__ uint4 sBx[2 * 8 * 2 * 32];   // 16 KB: bf16 B-frags of x
  __shared__ uint4 sBg[2 * 8 * 2 * 32];   // 16 KB: bf16 B-frags of g
  __shared__ float sBias[C_DIM], sGamma[C_DIM], sBeta[C_DIM], sWpsi[C_DIM];
  __shared__ float sRed1[2 * NTHREADS], sRed2[2 * NTHREADS];

  const int t = threadIdx.x;
  const int G = blockIdx.x;                 // 2048 groups of 32 pixels
  const int b  = G >> 7;                    // 128 groups per batch image
  const int rm = G & 127;
  const int h  = rm >> 1;
  const int w0 = (rm & 1) << 5;
  const size_t base = (size_t)b * BSTRIDE + (size_t)h * 64 + (size_t)w0;

  unsigned int* const sBxw = (unsigned int*)sBx;
  unsigned int* const sBgw = (unsigned int*)sBg;

  // ------- Phase 1: stage x,g into LDS, bf16, fragment-order layout -------
  for (int e = t; e < (C_DIM / 2) * PIXB; e += NTHREADS) {
    const int pi = e >> 5;          // channel-pair index (0..127)
    const int p  = e & 31;          // pixel in group (0..31)
    const int c0 = pi << 1;
    const float xv0 = x[base + (size_t)c0 * HW + p];
    const float xv1 = x[base + (size_t)(c0 + 1) * HW + p];
    const float gv0 = g[base + (size_t)c0 * HW + p];
    const float gv1 = g[base + (size_t)(c0 + 1) * HW + p];
    // fragment coordinates for K = c0 (pair c0,c0+1)
    const int v    = (pi & 3) | (((pi >> 3) & 1) << 2);  // dword within fragment
    const int kb   = pi >> 4;             // K-block (32 channels)
    const int hi   = (pi >> 2) & 1;       // lane half (K 0-7,16-23 vs 8-15,24-31)
    const int nt   = p >> 4;              // N-tile
    const int lane = hi * 16 + (p & 15);
    const int idx  = BFRAG_U4(nt, kb, v >> 2, lane) * 4 + (v & 3);
    sBxw[idx] = pk_bf16(xv0, xv1);
    sBgw[idx] = pk_bf16(gv0, gv1);
  }
  if (t < C_DIM) {
    sBias[t]  = bxg[t];
    sGamma[t] = ln_gamma[t];
    sBeta[t]  = ln_beta[t];
    sWpsi[t]  = Wpsi[t];
  }
  __syncthreads();

  // ---------------- Phase 2: WMMA GEMMs (K = 2*256 bf16, 2 N-tiles) -------
  const int wave = t >> 5;          // = M-tile (output-channel tile)
  const int lane = t & 31;
  const int p    = lane & 15;
  const int hi   = lane >> 4;

  const v8u* __restrict__ afrag = (const v8u*)ws;   // 32B fragments per lane

  v8f acc0 = {}, acc1 = {};
#pragma unroll
  for (int kb = 0; kb < 8; ++kb) {                 // 8 blocks of K=32
    // A-fragments: pre-packed, contiguous, coalesced (2x global_load_b128)
    const v8u ax = afrag[(wave * 8 + kb) * 32 + lane];
    const v8u ag = afrag[((16 + wave) * 8 + kb) * 32 + lane];
    // B-fragments: 2x ds_load_b128 each, straight into operand registers
    FragU bx0, bx1, bg0, bg1;
    bx0.q[0] = sBx[BFRAG_U4(0, kb, 0, lane)];
    bx0.q[1] = sBx[BFRAG_U4(0, kb, 1, lane)];
    bx1.q[0] = sBx[BFRAG_U4(1, kb, 0, lane)];
    bx1.q[1] = sBx[BFRAG_U4(1, kb, 1, lane)];
    bg0.q[0] = sBg[BFRAG_U4(0, kb, 0, lane)];
    bg0.q[1] = sBg[BFRAG_U4(0, kb, 1, lane)];
    bg1.q[0] = sBg[BFRAG_U4(1, kb, 0, lane)];
    bg1.q[1] = sBg[BFRAG_U4(1, kb, 1, lane)];
    // interleave accumulators so consecutive WMMAs are independent
    acc0 = __builtin_amdgcn_wmma_f32_16x16x32_bf16(
        false, __builtin_bit_cast(v16bf, ax), false, __builtin_bit_cast(v16bf, bx0.v),
        (short)0, acc0, false, false);
    acc1 = __builtin_amdgcn_wmma_f32_16x16x32_bf16(
        false, __builtin_bit_cast(v16bf, ax), false, __builtin_bit_cast(v16bf, bx1.v),
        (short)0, acc1, false, false);
    acc0 = __builtin_amdgcn_wmma_f32_16x16x32_bf16(
        false, __builtin_bit_cast(v16bf, ag), false, __builtin_bit_cast(v16bf, bg0.v),
        (short)0, acc0, false, false);
    acc1 = __builtin_amdgcn_wmma_f32_16x16x32_bf16(
        false, __builtin_bit_cast(v16bf, ag), false, __builtin_bit_cast(v16bf, bg1.v),
        (short)0, acc1, false, false);
  }

  // ---------------- Phase 3: bias + relu, LayerNorm stats over C ----------
  // D layout: lane l -> pixel (l&15) of its N-tile, VGPR r -> channel
  //           wave*16 + r + 8*(l>>4)
  const int chBase = wave * 16 + hi * 8;
  float att0[8], att1[8];
  float s0 = 0.f, sq0 = 0.f, s1 = 0.f, sq1 = 0.f;
#pragma unroll
  for (int r = 0; r < 8; ++r) {
    const float bias = sBias[chBase + r];
    float a0 = fmaxf(acc0[r] + bias, 0.f);
    float a1 = fmaxf(acc1[r] + bias, 0.f);
    att0[r] = a0; att1[r] = a1;
    s0 += a0; sq0 += a0 * a0;
    s1 += a1; sq1 += a1 * a1;
  }
  sRed1[t] = s0;  sRed1[NTHREADS + t] = s1;
  sRed2[t] = sq0; sRed2[NTHREADS + t] = sq1;
  __syncthreads();

  float S0 = 0.f, SQ0 = 0.f, S1 = 0.f, SQ1 = 0.f;
#pragma unroll
  for (int w = 0; w < NWAVES; ++w) {
    S0  += sRed1[w * 32 + p] + sRed1[w * 32 + 16 + p];
    SQ0 += sRed2[w * 32 + p] + sRed2[w * 32 + 16 + p];
    S1  += sRed1[NTHREADS + w * 32 + p] + sRed1[NTHREADS + w * 32 + 16 + p];
    SQ1 += sRed2[NTHREADS + w * 32 + p] + sRed2[NTHREADS + w * 32 + 16 + p];
  }
  const float mean0 = S0 * (1.f / 256.f);
  const float mean1 = S1 * (1.f / 256.f);
  const float rstd0 = rsqrtf(SQ0 * (1.f / 256.f) - mean0 * mean0 + 1e-5f);
  const float rstd1 = rsqrtf(SQ1 * (1.f / 256.f) - mean1 * mean1 + 1e-5f);

  // ---------------- Phase 4: psi projection + sigmoid ---------------------
  float ps0 = 0.f, ps1 = 0.f;
#pragma unroll
  for (int r = 0; r < 8; ++r) {
    const int ch = chBase + r;
    const float gm = sGamma[ch], bt = sBeta[ch], wp = sWpsi[ch];
    ps0 += ((att0[r] - mean0) * rstd0 * gm + bt) * wp;
    ps1 += ((att1[r] - mean1) * rstd1 * gm + bt) * wp;
  }
  __syncthreads();              // all reads of sRed1/sRed2 above complete
  sRed1[t] = ps0;  sRed1[NTHREADS + t] = ps1;
  __syncthreads();

  float PS0 = 0.f, PS1 = 0.f;
#pragma unroll
  for (int w = 0; w < NWAVES; ++w) {
    PS0 += sRed1[w * 32 + p] + sRed1[w * 32 + 16 + p];
    PS1 += sRed1[NTHREADS + w * 32 + p] + sRed1[NTHREADS + w * 32 + 16 + p];
  }
  const float bp = bpsi[0];
  const float sig0 = 1.f / (1.f + __expf(-(PS0 + bp)));
  const float sig1 = 1.f / (1.f + __expf(-(PS1 + bp)));

  // ---------------- Phase 5: gate original x (L2-hot), store --------------
#pragma unroll
  for (int r = 0; r < 8; ++r) {
    const int ch = chBase + r;
    const size_t i0 = base + (size_t)ch * HW + p;        // N-tile 0
    const size_t i1 = i0 + 16;                           // N-tile 1
    out[i0] = sig0 * x[i0];
    out[i1] = sig1 * x[i1];
  }
}

extern "C" void kernel_launch(void* const* d_in, const int* in_sizes, int n_in,
                              void* d_out, int out_size, void* d_ws, size_t ws_size,
                              hipStream_t stream) {
  (void)in_sizes; (void)n_in; (void)out_size; (void)ws_size;
  const float* x        = (const float*)d_in[0];
  const float* g        = (const float*)d_in[1];
  const float* Wx       = (const float*)d_in[2];
  const float* Wg       = (const float*)d_in[3];
  const float* Wpsi     = (const float*)d_in[4];
  const float* ln_gamma = (const float*)d_in[5];
  const float* ln_beta  = (const float*)d_in[6];
  const float* bxg      = (const float*)d_in[7];
  const float* bpsi     = (const float*)d_in[8];
  float* out = (float*)d_out;
  unsigned int* ws = (unsigned int*)d_ws;     // needs 256 KB

  // 1) pack Wx/Wg into bf16 WMMA A-fragments (64K uints)
  pack_weights_kernel<<<256, 256, 0, stream>>>(Wx, Wg, ws);
  // 2) fused attention gate: 65536 pixels / 32 per block = 2048 blocks
  attn_gate_kernel<<<2048, NTHREADS, 0, stream>>>(
      x, g, ws, Wpsi, ln_gamma, ln_beta, bxg, bpsi, out);
}
